// MultiHeadAttention_62130996904388
// MI455X (gfx1250) — matrix operand order repaired and checked by reading, not compiled, not run on hardware
//
#include <hip/hip_runtime.h>
#include <hip/hip_bf16.h>

// MHA forward for B=2, S=2048, D=1024, H=16, HD=64 on gfx1250 (MI455X).
// Pipeline: [qkv GEMM -> f16 Q/K/V in ws] -> [flash attention] -> [out proj GEMM].
// All matrix math via v_wmma_f32_16x16x32_f16 (fp32 accumulate).
// GEMMs are software-pipelined: next K-step is fetched from HBM into registers
// while the current K-step runs on the matrix units, then committed to LDS.

typedef __attribute__((ext_vector_type(16))) _Float16 v16h;
typedef __attribute__((ext_vector_type(8)))  _Float16 v8h;
typedef __attribute__((ext_vector_type(4)))  _Float16 v4h;
typedef __attribute__((ext_vector_type(8)))  float    v8f;

#define D_DIM   1024
#define S_DIM   2048
#define H_DIM   16
#define HD_DIM  64
#define M_DIM   4096   // B*S

// ---------------------------------------------------------------------------
// WMMA fragment loaders from LDS, per CDNA5 ISA 7.12.2 lane layouts (wave32).
// A (16x32 f16): lane L holds row M=L%16; lanes 0-15 hold K 0-7 & 16-23,
// lanes 16-31 hold K 8-15 & 24-31 -> two contiguous 8-half (16B) runs.
// B (32x16 f16): lane L holds col N=L%16; lanes 0-15 hold K 0-7 & 16-23,
// lanes 16-31 hold K 8-15 & 24-31 -> two 16B runs if the LDS tile is stored [N][K].
// C/D (16x16 f32): lane L holds col N=L%16; VGPR r holds row r (+8 for
// lanes 16-31).
// ---------------------------------------------------------------------------
__device__ __forceinline__ v16h frag_a_lds(const _Float16* base, int ld) {
  const int lane = threadIdx.x & 31;
  const int m  = lane & 15;
  const int kb = (lane & 16) >> 1;      // 0 or 8
  const _Float16* p = base + m * ld + kb;
  v8h lo = *(const v8h*)(p);
  v8h hi = *(const v8h*)(p + 16);
  v16h a;
#pragma unroll
  for (int i = 0; i < 8; ++i) { a[i] = lo[i]; a[i + 8] = hi[i]; }
  return a;
}

__device__ __forceinline__ v16h frag_b_lds(const _Float16* base, int ld) {
  // base points at Bs[n0][k0] of an LDS tile stored transposed as [N][K]
  const int lane = threadIdx.x & 31;
  const int n  = lane & 15;
  const int kb = (lane & 16) >> 1;      // 0 or 8
  const _Float16* p = base + n * ld + kb;
  v8h lo = *(const v8h*)(p);
  v8h hi = *(const v8h*)(p + 16);
  v16h b;
#pragma unroll
  for (int i = 0; i < 8; ++i) { b[i] = lo[i]; b[i + 8] = hi[i]; }
  return b;
}

__device__ __forceinline__ v4h cvt4(const float4 v) {
  v4h h;
  h[0] = (_Float16)v.x; h[1] = (_Float16)v.y;
  h[2] = (_Float16)v.z; h[3] = (_Float16)v.w;
  return h;
}

// ---------------------------------------------------------------------------
// Kernel 1: Q/K/V projection. Block tile 128(M) x 256(N), 256 threads
// (8 waves in 2x4), wave tile 64x64 = 4x4 WMMA tiles -> 16 wmma / K-step
// with only 8 fragment loads (4x reuse of A and B frags). K-step 32.
// Software pipelined: global->regs for step k+1 issued before WMMA of step k.
// blockIdx.z selects W/bias/dst. Output layout [B,H,S,HD] f16.
// ---------------------------------------------------------------------------
__global__ __launch_bounds__(256)
void qkv_gemm_kernel(const float* __restrict__ x,
                     const float* __restrict__ Wq, const float* __restrict__ bq,
                     const float* __restrict__ Wk, const float* __restrict__ bk,
                     const float* __restrict__ Wv, const float* __restrict__ bv,
                     _Float16* __restrict__ Qh, _Float16* __restrict__ Kh,
                     _Float16* __restrict__ Vh) {
  __shared__ __align__(16) _Float16 As[128][40];   // [M][K] row-major (pad 8)
  __shared__ __align__(16) _Float16 Bs[256][40];   // [N][K] transposed (pad 8)

  const float* W;
  const float* bias;
  _Float16*    dst;
  if (blockIdx.z == 0)      { W = Wq; bias = bq; dst = Qh; }
  else if (blockIdx.z == 1) { W = Wk; bias = bk; dst = Kh; }
  else                      { W = Wv; bias = bv; dst = Vh; }

  const int n0 = blockIdx.x * 256;
  const int m0 = blockIdx.y * 128;
  const int t    = threadIdx.x;
  const int lane = t & 31;
  const int w    = t >> 5;
  const int wm   = (w >> 2) * 64;   // wave row base within 128
  const int wn   = (w & 3) * 64;    // wave col base within 256

  // staging indices
  const int ac  = (t & 7) * 4;      // A col (0..28)
  const int ar  = t >> 3;           // A row base (0..31), rows ar + i*32
  const int bn  = (t & 63) * 4;     // B col (0..252)
  const int bk0 = t >> 6;           // B k base (0..3), k = bk0 + i*4

  v8f acc[4][4] = {};
  float4 ra[4], rb[8];

  // ---- prologue: stage K-step 0 ----
#pragma unroll
  for (int i = 0; i < 4; ++i)
    ra[i] = *(const float4*)(&x[(size_t)(m0 + ar + i * 32) * D_DIM + ac]);
#pragma unroll
  for (int i = 0; i < 8; ++i)
    rb[i] = *(const float4*)(&W[(size_t)(bk0 + i * 4) * D_DIM + n0 + bn]);
#pragma unroll
  for (int i = 0; i < 4; ++i)
    *(v4h*)(&As[ar + i * 32][ac]) = cvt4(ra[i]);
#pragma unroll
  for (int i = 0; i < 8; ++i) {
    const v4h h = cvt4(rb[i]);
    const int kk = bk0 + i * 4;
#pragma unroll
    for (int j = 0; j < 4; ++j) Bs[bn + j][kk] = h[j];
  }
  __syncthreads();

  for (int k0 = 0; k0 < D_DIM; k0 += 32) {
    const bool more = (k0 + 32) < D_DIM;
    if (more) {   // fetch next K-step into registers (overlaps with WMMA below)
      const int kn = k0 + 32;
#pragma unroll
      for (int i = 0; i < 4; ++i)
        ra[i] = *(const float4*)(&x[(size_t)(m0 + ar + i * 32) * D_DIM + kn + ac]);
#pragma unroll
      for (int i = 0; i < 8; ++i)
        rb[i] = *(const float4*)(&W[(size_t)(kn + bk0 + i * 4) * D_DIM + n0 + bn]);
      __builtin_prefetch(&x[(size_t)(m0 + ar) * D_DIM + kn + 32 + ac], 0, 1);
      __builtin_prefetch(&W[(size_t)(kn + 32 + bk0) * D_DIM + n0 + bn], 0, 1);
    }

    // compute current K-step from LDS
    v16h afrag[4], bfrag[4];
#pragma unroll
    for (int i = 0; i < 4; ++i) afrag[i] = frag_a_lds(&As[wm + i * 16][0], 40);
#pragma unroll
    for (int j = 0; j < 4; ++j) bfrag[j] = frag_b_lds(&Bs[wn + j * 16][0], 40);
#pragma unroll
    for (int i = 0; i < 4; ++i)
#pragma unroll
      for (int j = 0; j < 4; ++j)
        acc[i][j] = __builtin_amdgcn_wmma_f32_16x16x32_f16(
            false, afrag[i], false, bfrag[j], (short)0, acc[i][j], false, false);

    if (more) {   // commit staged registers to LDS
      __syncthreads();
#pragma unroll
      for (int i = 0; i < 4; ++i)
        *(v4h*)(&As[ar + i * 32][ac]) = cvt4(ra[i]);
#pragma unroll
      for (int i = 0; i < 8; ++i) {
        const v4h h = cvt4(rb[i]);
        const int kk = bk0 + i * 4;
#pragma unroll
        for (int j = 0; j < 4; ++j) Bs[bn + j][kk] = h[j];
      }
      __syncthreads();
    }
  }

  // Epilogue: bias add, f16 convert, scatter to [B,H,S,HD]
  const int nl    = lane & 15;
  const int rbase = (lane & 16) >> 1;
#pragma unroll
  for (int i = 0; i < 4; ++i) {
#pragma unroll
    for (int j = 0; j < 4; ++j) {
      const int   gn  = n0 + wn + j * 16 + nl;
      const float bsv = bias[gn];
      const int   h   = gn >> 6;
      const int   hd  = gn & 63;
#pragma unroll
      for (int r = 0; r < 8; ++r) {
        const int gm = m0 + wm + i * 16 + rbase + r;
        const int bb = gm >> 11;
        const int s  = gm & 2047;
        dst[((size_t)((bb * H_DIM + h) * S_DIM + s)) * HD_DIM + hd] =
            (_Float16)(acc[i][j][r] + bsv);
      }
    }
  }
}

// ---------------------------------------------------------------------------
// Kernel 2: flash attention. One block per (b,h, 64-query chunk).
// O accumulator lives in WMMA C fragments; online softmax with per-row
// rescale. 64-key chunks, V staged transposed so PV B-frags are contiguous.
// ---------------------------------------------------------------------------
__global__ __launch_bounds__(256)
void attention_kernel(const _Float16* __restrict__ Qh,
                      const _Float16* __restrict__ Kh,
                      const _Float16* __restrict__ Vh,
                      _Float16* __restrict__ Ah) {
  __shared__ __align__(16) _Float16 Qs[64][72];    // [q][d]
  __shared__ __align__(16) _Float16 Ks[64][72];    // [key][d]  (== [N][K] for QK^T)
  __shared__ __align__(16) _Float16 Vst[64][72];   // [d][key]  (== [N][K] for PV)
  __shared__ __align__(16) _Float16 Ps[64][72];    // probabilities f16
  __shared__ __align__(16) float    Ss[64][68];    // fp32 score tile
  __shared__ float mrow[64], lrow[64], arow[64];

  const int    bh   = blockIdx.y;                  // 0..31  (b*16 + h)
  const int    q0   = blockIdx.x * 64;
  const size_t base = (size_t)bh * S_DIM * HD_DIM;

  const int t     = threadIdx.x;
  const int lane  = t & 31;
  const int w     = t >> 5;
  const int wr    = (w >> 1) * 16;   // wave row-tile base within 64
  const int wc    = (w & 1) * 32;    // wave col base within 64 (2 tiles)
  const int nl    = lane & 15;
  const int rbase = (lane & 16) >> 1;

  const int c8 = (t & 7) * 8;
  const int r0 = t >> 3;

  {   // load Q chunk once
#pragma unroll
    for (int i = 0; i < 2; ++i) {
      const int row = r0 + i * 32;
      *(v8h*)(&Qs[row][c8]) = *(const v8h*)(&Qh[base + (size_t)(q0 + row) * HD_DIM + c8]);
    }
  }
  if (t < 64) { mrow[t] = -3.0e38f; lrow[t] = 0.0f; arow[t] = 1.0f; }

  v8f oacc[2] = {};

  for (int j0 = 0; j0 < S_DIM; j0 += 64) {
    __syncthreads();
    {   // load K chunk [key][d] and V chunk transposed [d][key]
#pragma unroll
      for (int i = 0; i < 2; ++i) {
        const int row = r0 + i * 32;
        *(v8h*)(&Ks[row][c8]) = *(const v8h*)(&Kh[base + (size_t)(j0 + row) * HD_DIM + c8]);
        v8h vv = *(const v8h*)(&Vh[base + (size_t)(j0 + row) * HD_DIM + c8]);
#pragma unroll
        for (int e = 0; e < 8; ++e) Vst[c8 + e][row] = vv[e];
      }
    }
    if (j0 + 64 < S_DIM) {   // global_prefetch_b8 of next K/V chunk
      __builtin_prefetch(&Kh[base + (size_t)(j0 + 64 + r0) * HD_DIM + c8], 0, 1);
      __builtin_prefetch(&Vh[base + (size_t)(j0 + 64 + r0) * HD_DIM + c8], 0, 1);
    }
    __syncthreads();

    // S = (Q K^T) * 1/sqrt(HD); wave computes rows [wr,wr+16), cols [wc,wc+32)
    {
      v8f sacc[2] = {};
#pragma unroll
      for (int kk = 0; kk < HD_DIM; kk += 32) {
        v16h a = frag_a_lds(&Qs[wr][kk], 72);
#pragma unroll
        for (int c = 0; c < 2; ++c) {
          v16h b = frag_b_lds(&Ks[wc + c * 16][kk], 72);
          sacc[c] = __builtin_amdgcn_wmma_f32_16x16x32_f16(
              false, a, false, b, (short)0, sacc[c], false, false);
        }
      }
#pragma unroll
      for (int c = 0; c < 2; ++c)
#pragma unroll
        for (int r = 0; r < 8; ++r)
          Ss[wr + rbase + r][wc + c * 16 + nl] = sacc[c][r] * 0.125f;
    }
    __syncthreads();

    // Online softmax row update (one thread per query row)
    if (t < 64) {
      const float mold = mrow[t];
      float mx = mold;
#pragma unroll 8
      for (int jj = 0; jj < 64; ++jj) mx = fmaxf(mx, Ss[t][jj]);
      const float alpha = __expf(mold - mx);
      float lsum = lrow[t] * alpha;
#pragma unroll 8
      for (int jj = 0; jj < 64; ++jj) {
        const float p = __expf(Ss[t][jj] - mx);
        Ps[t][jj] = (_Float16)p;
        lsum += p;
      }
      mrow[t] = mx; lrow[t] = lsum; arow[t] = alpha;
    }
    __syncthreads();

    // Rescale running O by alpha, then O += P * V
#pragma unroll
    for (int c = 0; c < 2; ++c)
#pragma unroll
      for (int r = 0; r < 8; ++r) oacc[c][r] *= arow[wr + rbase + r];
#pragma unroll
    for (int kk = 0; kk < 64; kk += 32) {
      v16h a = frag_a_lds(&Ps[wr][kk], 72);
#pragma unroll
      for (int c = 0; c < 2; ++c) {
        v16h b = frag_b_lds(&Vst[wc + c * 16][kk], 72);
        oacc[c] = __builtin_amdgcn_wmma_f32_16x16x32_f16(
            false, a, false, b, (short)0, oacc[c], false, false);
      }
    }
  }
  __syncthreads();

  // Normalize and store attn output directly in [B,S,D] layout (f16)
  const int bb = bh >> 4;
  const int h  = bh & 15;
#pragma unroll
  for (int c = 0; c < 2; ++c) {
#pragma unroll
    for (int r = 0; r < 8; ++r) {
      const int   row = wr + rbase + r;
      const float inv = 1.0f / lrow[row];
      const int   gq  = q0 + row;
      Ah[((size_t)(bb * S_DIM + gq)) * D_DIM + h * HD_DIM + wc + c * 16 + nl] =
          (_Float16)(oacc[c][r] * inv);
    }
  }
}

// ---------------------------------------------------------------------------
// Kernel 3: output projection. Same pipelined skeleton as kernel 1; f16 A
// from workspace, fp32 W converted on stage, fp32 output + bias to d_out.
// ---------------------------------------------------------------------------
__global__ __launch_bounds__(256)
void out_proj_kernel(const _Float16* __restrict__ Ah,
                     const float* __restrict__ Wo, const float* __restrict__ bo,
                     float* __restrict__ out) {
  __shared__ __align__(16) _Float16 As[128][40];
  __shared__ __align__(16) _Float16 Bs[256][40];

  const int n0 = blockIdx.x * 256;
  const int m0 = blockIdx.y * 128;
  const int t    = threadIdx.x;
  const int lane = t & 31;
  const int w    = t >> 5;
  const int wm   = (w >> 2) * 64;
  const int wn   = (w & 3) * 64;

  const int ac  = (t & 3) * 8;      // A col (halves)
  const int ar  = t >> 2;           // A row base (0..63), rows ar + i*64
  const int bn  = (t & 63) * 4;
  const int bk0 = t >> 6;

  v8f acc[4][4] = {};
  v8h    ra[2];
  float4 rb[8];

  // ---- prologue ----
#pragma unroll
  for (int i = 0; i < 2; ++i)
    ra[i] = *(const v8h*)(&Ah[(size_t)(m0 + ar + i * 64) * D_DIM + ac]);
#pragma unroll
  for (int i = 0; i < 8; ++i)
    rb[i] = *(const float4*)(&Wo[(size_t)(bk0 + i * 4) * D_DIM + n0 + bn]);
#pragma unroll
  for (int i = 0; i < 2; ++i)
    *(v8h*)(&As[ar + i * 64][ac]) = ra[i];
#pragma unroll
  for (int i = 0; i < 8; ++i) {
    const v4h h = cvt4(rb[i]);
    const int kk = bk0 + i * 4;
#pragma unroll
    for (int j = 0; j < 4; ++j) Bs[bn + j][kk] = h[j];
  }
  __syncthreads();

  for (int k0 = 0; k0 < D_DIM; k0 += 32) {
    const bool more = (k0 + 32) < D_DIM;
    if (more) {
      const int kn = k0 + 32;
#pragma unroll
      for (int i = 0; i < 2; ++i)
        ra[i] = *(const v8h*)(&Ah[(size_t)(m0 + ar + i * 64) * D_DIM + kn + ac]);
#pragma unroll
      for (int i = 0; i < 8; ++i)
        rb[i] = *(const float4*)(&Wo[(size_t)(kn + bk0 + i * 4) * D_DIM + n0 + bn]);
      __builtin_prefetch(&Ah[(size_t)(m0 + ar) * D_DIM + kn + 32 + ac], 0, 1);
      __builtin_prefetch(&Wo[(size_t)(kn + 32 + bk0) * D_DIM + n0 + bn], 0, 1);
    }

    v16h afrag[4], bfrag[4];
#pragma unroll
    for (int i = 0; i < 4; ++i) afrag[i] = frag_a_lds(&As[wm + i * 16][0], 40);
#pragma unroll
    for (int j = 0; j < 4; ++j) bfrag[j] = frag_b_lds(&Bs[wn + j * 16][0], 40);
#pragma unroll
    for (int i = 0; i < 4; ++i)
#pragma unroll
      for (int j = 0; j < 4; ++j)
        acc[i][j] = __builtin_amdgcn_wmma_f32_16x16x32_f16(
            false, afrag[i], false, bfrag[j], (short)0, acc[i][j], false, false);

    if (more) {
      __syncthreads();
#pragma unroll
      for (int i = 0; i < 2; ++i)
        *(v8h*)(&As[ar + i * 64][ac]) = ra[i];
#pragma unroll
      for (int i = 0; i < 8; ++i) {
        const v4h h = cvt4(rb[i]);
        const int kk = bk0 + i * 4;
#pragma unroll
        for (int j = 0; j < 4; ++j) Bs[bn + j][kk] = h[j];
      }
      __syncthreads();
    }
  }

  const int nl    = lane & 15;
  const int rbase = (lane & 16) >> 1;
#pragma unroll
  for (int i = 0; i < 4; ++i) {
#pragma unroll
    for (int j = 0; j < 4; ++j) {
      const int   gn  = n0 + wn + j * 16 + nl;
      const float bsv = bo[gn];
#pragma unroll
      for (int r = 0; r < 8; ++r) {
        const int gm = m0 + wm + i * 16 + rbase + r;
        out[(size_t)gm * D_DIM + gn] = acc[i][j][r] + bsv;
      }
    }
  }
}

// ---------------------------------------------------------------------------
extern "C" void kernel_launch(void* const* d_in, const int* in_sizes, int n_in,
                              void* d_out, int out_size, void* d_ws, size_t ws_size,
                              hipStream_t stream) {
  const float* x  = (const float*)d_in[0];
  const float* Wq = (const float*)d_in[1];
  const float* bq = (const float*)d_in[2];
  const float* Wk = (const float*)d_in[3];
  const float* bk = (const float*)d_in[4];
  const float* Wv = (const float*)d_in[5];
  const float* bv = (const float*)d_in[6];
  const float* Wo = (const float*)d_in[7];
  const float* bo = (const float*)d_in[8];
  float* out = (float*)d_out;

  // Workspace layout: 4 x 8MB f16 buffers (Q, K, V in [B,H,S,HD]; attn in [B,S,D])
  char* ws = (char*)d_ws;
  const size_t BUF = (size_t)M_DIM * D_DIM * sizeof(_Float16);   // 8 MB
  _Float16* Qh = (_Float16*)(ws + 0 * BUF);
  _Float16* Kh = (_Float16*)(ws + 1 * BUF);
  _Float16* Vh = (_Float16*)(ws + 2 * BUF);
  _Float16* Ah = (_Float16*)(ws + 3 * BUF);

  qkv_gemm_kernel<<<dim3(D_DIM / 256, M_DIM / 128, 3), 256, 0, stream>>>(
      x, Wq, bq, Wk, bk, Wv, bv, Qh, Kh, Vh);
  attention_kernel<<<dim3(S_DIM / 64, 32), 256, 0, stream>>>(Qh, Kh, Vh, Ah);
  out_proj_kernel<<<dim3(D_DIM / 256, M_DIM / 128), 256, 0, stream>>>(Ah, Wo, bo, out);
}
